// RestoreDiNAT_29729763623683
// MI455X (gfx1250) — matrix operand order repaired and checked by reading, not compiled, not run on hardware
//
#include <hip/hip_runtime.h>

typedef __attribute__((ext_vector_type(16))) _Float16 v16h;
typedef __attribute__((ext_vector_type(4)))  _Float16 v4h;
typedef __attribute__((ext_vector_type(8)))  float    v8f;

union HFrag { v16h v; unsigned u[8]; };

// =====================================================================
// GEMM: Y[N,P] = Wm[N,K] * X[K,P] (+ bias[N]), WMMA f32_16x16x32_f16.
// Block: 256 threads (8 wave32), tile 32 out-channels x 256 pixels.
// Each wave: 1 A fragment x 4 B fragments -> 4 wmma per K-step.
// Software-pipelined: double-buffered LDS, next tile's global loads are
// issued before the current tile's wmma ops; one barrier per K-step.
// P must be a multiple of 256 (always true: HW in {256,1024,4096,16384}).
// =====================================================================
template<bool EDGE, bool HASBIAS>
__global__ void __launch_bounds__(256)
k_gemm_wmma(const float* __restrict__ X, const float* __restrict__ Wm,
            const float* __restrict__ bias, float* __restrict__ Y,
            int N, int Kd, int P)
{
  __shared__ _Float16 sW[2][32][40];    // [buf][outch][k]
  __shared__ _Float16 sX[2][256][40];   // [buf][pixel][k] (transposed stage)
  const int n0   = blockIdx.x * 32;
  const int p0   = blockIdx.y * 256;
  const int tid  = threadIdx.x;
  const int lane = tid & 31;
  const int wave = tid >> 5;
  const int tR   = wave & 1;   // out-channel sub-tile (0/1)
  const int tCq  = wave >> 1;  // pixel quad (0..3): covers col-tiles tCq*4..tCq*4+3
  const int kg   = lane >> 4;
  const int row  = lane & 15;

  v8f acc[4] = {{}, {}, {}, {}};

  float4 xreg[8];
  float4 wreg;
  float  wregS[4];

  auto load_tile = [&](int k0) {
    if (EDGE) {
#pragma unroll
      for (int i = 0; i < 4; ++i) {
        int e = tid + i * 256;
        int r = e >> 5, c = e & 31;
        int rr = n0 + r, cc = k0 + c;
        int rcl = rr < N ? rr : N - 1;
        int ccl = cc < Kd ? cc : Kd - 1;
        float v = Wm[(size_t)rcl * Kd + ccl];
        wregS[i] = (rr < N && cc < Kd) ? v : 0.f;
      }
    } else {
      int r  = tid >> 3;
      int c4 = (tid & 7) * 4;
      wreg = *(const float4*)&Wm[(size_t)(n0 + r) * Kd + (k0 + c4)];
    }
#pragma unroll
    for (int i = 0; i < 8; ++i) {
      int e  = tid + i * 256;
      int kk = e >> 6;
      int p4 = (e & 63) * 4;
      if (EDGE) {
        int kkg = k0 + kk;
        int kcl = kkg < Kd ? kkg : Kd - 1;
        float4 vv = *(const float4*)&X[(size_t)kcl * P + p0 + p4];
        if (kkg >= Kd) { vv.x = 0.f; vv.y = 0.f; vv.z = 0.f; vv.w = 0.f; }
        xreg[i] = vv;
      } else {
        xreg[i] = *(const float4*)&X[(size_t)(k0 + kk) * P + p0 + p4];
      }
    }
  };

  auto store_tile = [&](int buf) {
    if (EDGE) {
#pragma unroll
      for (int i = 0; i < 4; ++i) {
        int e = tid + i * 256;
        int r = e >> 5, c = e & 31;
        sW[buf][r][c] = (_Float16)wregS[i];
      }
    } else {
      int r  = tid >> 3;
      int c4 = (tid & 7) * 4;
      v4h hw;
      hw[0] = (_Float16)wreg.x; hw[1] = (_Float16)wreg.y;
      hw[2] = (_Float16)wreg.z; hw[3] = (_Float16)wreg.w;
      *(v4h*)&sW[buf][r][c4] = hw;
    }
#pragma unroll
    for (int i = 0; i < 8; ++i) {
      int e  = tid + i * 256;
      int kk = e >> 6;
      int p4 = (e & 63) * 4;
      sX[buf][p4 + 0][kk] = (_Float16)xreg[i].x;
      sX[buf][p4 + 1][kk] = (_Float16)xreg[i].y;
      sX[buf][p4 + 2][kk] = (_Float16)xreg[i].z;
      sX[buf][p4 + 3][kk] = (_Float16)xreg[i].w;
    }
  };

  const int kiters = (Kd + 31) >> 5;

  load_tile(0);
  store_tile(0);
  __syncthreads();

  for (int kt = 0; kt < kiters; ++kt) {
    const int  cur  = kt & 1;
    const bool more = (kt + 1) < kiters;
    if (more) load_tile((kt + 1) * 32);   // issue global loads early (latency hidden)

    // A fragment (16x32 f16): VGPR0-3: K 0..7, VGPR4-7: K 16..23, +8 per lane-half
    HFrag a;
#pragma unroll
    for (int t = 0; t < 8; ++t) {
      int kb = ((t < 4) ? (2 * t) : (16 + 2 * (t - 4))) + 8 * kg;
      a.u[t] = *(const unsigned*)&sW[cur][tR * 16 + row][kb];
    }
    // 4 B fragments (32x16 f16 each): lane = column, VGPR0-3: K 0..7, VGPR4-7: K 16..23, +8 per lane-half
#pragma unroll
    for (int s = 0; s < 4; ++s) {
      HFrag b;
      const int coltile = tCq * 4 + s;
#pragma unroll
      for (int t = 0; t < 8; ++t)
        b.u[t] = *(const unsigned*)&sX[cur][coltile * 16 + row][((t < 4) ? (2 * t) : (16 + 2 * (t - 4))) + 8 * kg];
      acc[s] = __builtin_amdgcn_wmma_f32_16x16x32_f16(false, a.v, false, b.v,
                                                      (short)0, acc[s], false, false);
    }

    if (more) store_tile(cur ^ 1);        // convert + stage into other buffer
    __syncthreads();
  }

  // hoist bias values for the 8 D-rows this lane owns (one batched load group)
  float bv[8];
  if (HASBIAS) {
#pragma unroll
    for (int r = 0; r < 8; ++r) {
      int mch = n0 + tR * 16 + r + 8 * kg;
      bv[r] = bias[EDGE ? (mch < N ? mch : N - 1) : mch];
    }
  }

#pragma unroll
  for (int s = 0; s < 4; ++s) {
    const int pix = p0 + (tCq * 4 + s) * 16 + row;   // D: col = lane&15
#pragma unroll
    for (int r = 0; r < 8; ++r) {                    // D: row = r + 8*(lane>>4)
      int mch = n0 + tR * 16 + r + 8 * kg;
      float o = acc[s][r];
      if (HASBIAS) o += bv[r];
      if (!EDGE || mch < N) Y[(size_t)mch * P + pix] = o;
    }
  }
}

// =====================================================================
// Dense conv 3x3, pad 1, NCHW / OIHW, optional add source. Branchless taps.
// =====================================================================
__global__ void k_conv3x3(const float* __restrict__ in, const float* __restrict__ w,
                          float* __restrict__ out, int Ci, int Co, int H, int W,
                          const float* __restrict__ addsrc)
{
  int HW = H * W;
  int i = blockIdx.x * blockDim.x + threadIdx.x;
  if (i >= Co * HW) return;
  int co = i / HW, p = i % HW, y = p / W, x = p % W;
  float acc = 0.f;
  for (int ci = 0; ci < Ci; ++ci) {
    const float* ip = in + (size_t)ci * HW;
    const float* wp = w + ((size_t)co * Ci + ci) * 9;
#pragma unroll
    for (int ky = 0; ky < 3; ++ky) {
#pragma unroll
      for (int kx = 0; kx < 3; ++kx) {
        int yy = y + ky - 1, xx = x + kx - 1;
        bool ok = ((unsigned)yy < (unsigned)H) && ((unsigned)xx < (unsigned)W);
        int yc = ok ? yy : y, xc = ok ? xx : x;
        float pv = ip[yc * W + xc];
        acc += (ok ? pv : 0.f) * wp[ky * 3 + kx];
      }
    }
  }
  if (addsrc) acc += addsrc[i];
  out[i] = acc;
}

// Depthwise conv 3x3 pad 1 (groups == channels). Branchless taps.
__global__ void k_dwconv3x3(const float* __restrict__ in, const float* __restrict__ w,
                            float* __restrict__ out, int C, int H, int W)
{
  int HW = H * W;
  int i = blockIdx.x * blockDim.x + threadIdx.x;
  if (i >= C * HW) return;
  int c = i / HW, p = i % HW, y = p / W, x = p % W;
  const float* ip = in + (size_t)c * HW;
  const float* wp = w + (size_t)c * 9;
  float acc = 0.f;
#pragma unroll
  for (int ky = 0; ky < 3; ++ky) {
#pragma unroll
    for (int kx = 0; kx < 3; ++kx) {
      int yy = y + ky - 1, xx = x + kx - 1;
      bool ok = ((unsigned)yy < (unsigned)H) && ((unsigned)xx < (unsigned)W);
      int yc = ok ? yy : y, xc = ok ? xx : x;
      float pv = ip[yc * W + xc];
      acc += (ok ? pv : 0.f) * wp[ky * 3 + kx];
    }
  }
  out[i] = acc;
}

// =====================================================================
// LayerNorm over channel dim, x/y layout [C, HW].
// =====================================================================
__global__ void k_layernorm(const float* __restrict__ x, const float* __restrict__ w,
                            const float* __restrict__ b, float* __restrict__ y,
                            int C, int HW)
{
  int p = blockIdx.x * blockDim.x + threadIdx.x;
  if (p >= HW) return;
  float mu = 0.f;
  for (int c = 0; c < C; ++c) mu += x[(size_t)c * HW + p];
  mu /= (float)C;
  float var = 0.f;
  for (int c = 0; c < C; ++c) { float t = x[(size_t)c * HW + p] - mu; var += t * t; }
  var /= (float)C;
  float inv = rsqrtf(var + 1e-5f);
  for (int c = 0; c < C; ++c)
    y[(size_t)c * HW + p] = (x[(size_t)c * HW + p] - mu) * inv * w[c] + b[c];
}

// =====================================================================
// Neighborhood attention (7x7 taps, dilation d), one thread = (pixel, head).
// Online (flash-style) softmax: single rolled pass over taps, no logit array.
// qkv layout: [3C, HW]; out: [C, HW]; rpb: [nh, 13, 13].
// =====================================================================
__device__ __forceinline__ int win_start(int idx, int len, int d)
{
  if (d <= 1) {
    int v = idx - 3; v = v < 0 ? 0 : v;
    int mx = len - 7; return v > mx ? mx : v;
  }
  int ni = idx - 3 * d;
  int imodd = idx % d;
  int a = (len / d) * d;
  int b = len - a;
  int upper = (imodd < b) ? (a + imodd - 6 * d) : (a + imodd - 7 * d);
  if (ni < 0) return imodd;
  return (idx + 3 * d >= len) ? upper : ni;
}

__device__ __forceinline__ int pbias_start(int idx, int len, int d)
{
  if (d <= 1) {
    int r = 3;
    if (idx < 3) r += 3 - idx;
    if (idx + 3 >= len) r += len - idx - 1 - 3;
    return r;
  }
  if (idx - 3 * d < 0) return 6 - idx / d;
  if (idx + 3 * d >= len) return (len - idx - 1) / d;
  return 3;
}

template<int DH>
__global__ void k_na2d(const float* __restrict__ qkv, const float* __restrict__ rpb,
                       float* __restrict__ out, int C, int nh, int H, int W, int d)
{
  int HW = H * W;
  int i = blockIdx.x * blockDim.x + threadIdx.x;
  if (i >= HW * nh) return;
  int h = i / HW, p = i % HW, py = p / W, px = p % W;

  int wsy = win_start(py, H, d), wsx = win_start(px, W, d);
  int pby = pbias_start(py, H, d), pbx = pbias_start(px, W, d);

  const float* q = qkv + (size_t)(h * DH) * HW;
  const float* k = qkv + (size_t)(C + h * DH) * HW;
  const float* v = qkv + (size_t)(2 * C + h * DH) * HW;
  const float* bp = rpb + (size_t)h * 13 * 13;

  float scale = rsqrtf((float)DH);
  float qr[DH];
#pragma unroll
  for (int c = 0; c < DH; ++c) qr[c] = q[(size_t)c * HW + p] * scale;

  float m = -3.0e38f, s = 0.f;
  float acc[DH];
#pragma unroll
  for (int c = 0; c < DH; ++c) acc[c] = 0.f;

#pragma unroll 1
  for (int ii = 0; ii < 7; ++ii) {
    int ny  = wsy + ii * d;
    int byo = (pby + ii) * 13 + pbx;
#pragma unroll 1
    for (int jj = 0; jj < 7; ++jj) {
      int nb = ny * W + (wsx + jj * d);
      float dot = 0.f;
#pragma unroll
      for (int c = 0; c < DH; ++c) dot += qr[c] * k[(size_t)c * HW + nb];
      dot += bp[byo + jj];
      float mn   = fmaxf(m, dot);
      float corr = expf(m - mn);      // 1.0 when no new max
      float w    = expf(dot - mn);
      s = s * corr + w;
#pragma unroll
      for (int c = 0; c < DH; ++c)
        acc[c] = acc[c] * corr + w * v[(size_t)c * HW + nb];
      m = mn;
    }
  }
  float inv = 1.f / s;
#pragma unroll
  for (int c = 0; c < DH; ++c) out[(size_t)(h * DH + c) * HW + p] = acc[c] * inv;
}

// =====================================================================
// Elementwise / layout kernels.
// =====================================================================
__global__ void k_add(float* __restrict__ x, const float* __restrict__ dlt, int n)
{
  int i = blockIdx.x * blockDim.x + threadIdx.x;
  if (i < n) x[i] += dlt[i];
}

__global__ void k_gate(const float* __restrict__ u, float* __restrict__ g, int hf, int HW)
{
  int i = blockIdx.x * blockDim.x + threadIdx.x;
  if (i >= hf * HW) return;
  int c = i / HW, p = i % HW;
  float x1 = u[(size_t)c * HW + p];
  float x2 = u[(size_t)(c + hf) * HW + p];
  float ge = 0.5f * x1 * (1.f + erff(x1 * 0.70710678118654752f));
  g[i] = ge * x2;
}

__global__ void k_unshuffle2(const float* __restrict__ in, float* __restrict__ out,
                             int C, int H, int W)
{
  int Ho = H / 2, Wo = W / 2;
  int n = 4 * C * Ho * Wo;
  int i = blockIdx.x * blockDim.x + threadIdx.x;
  if (i >= n) return;
  int wo = i % Wo; int t = i / Wo; int ho = t % Ho; int oc = t / Ho;
  int c = oc >> 2, a = (oc >> 1) & 1, b = oc & 1;
  out[i] = in[((size_t)c * H + (ho * 2 + a)) * W + (wo * 2 + b)];
}

__global__ void k_shuffle2(const float* __restrict__ in, float* __restrict__ out,
                           int C, int H, int W)
{
  int Co = C / 4, Ho = 2 * H, Wo = 2 * W;
  int n = Co * Ho * Wo;
  int i = blockIdx.x * blockDim.x + threadIdx.x;
  if (i >= n) return;
  int wo = i % Wo; int t = i / Wo; int ho = t % Ho; int oc = t / Ho;
  int a = ho & 1, b = wo & 1;
  out[i] = in[((size_t)(oc * 4 + a * 2 + b) * H + (ho >> 1)) * W + (wo >> 1)];
}

// =====================================================================
// Host orchestration.
// =====================================================================
struct TransP {
  const float *n1w, *n1b, *qkv_w, *qkv_b, *rpb, *proj_w, *proj_b,
              *n2w, *n2b, *pin, *dw, *pout;
};

static inline void launch_gemm(hipStream_t s, const float* X, const float* Wm,
                               const float* bias, float* Y, int N, int Kd, int P)
{
  dim3 g((N + 31) / 32, P / 256);
  bool edge = ((N & 31) != 0) || ((Kd & 31) != 0);
  if (bias) {
    if (edge) k_gemm_wmma<true , true ><<<g, 256, 0, s>>>(X, Wm, bias, Y, N, Kd, P);
    else      k_gemm_wmma<false, true ><<<g, 256, 0, s>>>(X, Wm, bias, Y, N, Kd, P);
  } else {
    if (edge) k_gemm_wmma<true , false><<<g, 256, 0, s>>>(X, Wm, bias, Y, N, Kd, P);
    else      k_gemm_wmma<false, false><<<g, 256, 0, s>>>(X, Wm, bias, Y, N, Kd, P);
  }
}

static void trans_block(hipStream_t s, float* x, const TransP& t,
                        int C, int nh, int d, int H, int W, int hf,
                        float* ybuf, float* attbuf, float* regA, float* regB)
{
  int HW = H * W;
  int pb = (HW + 255) / 256;
  int nb = (C * HW + 255) / 256;

  // --- attention branch ---
  k_layernorm<<<pb, 256, 0, s>>>(x, t.n1w, t.n1b, ybuf, C, HW);
  launch_gemm(s, ybuf, t.qkv_w, t.qkv_b, regA, 3 * C, C, HW);
  int dh = C / nh;
  int tb = (HW * nh + 255) / 256;
  if (dh == 32) k_na2d<32><<<tb, 256, 0, s>>>(regA, t.rpb, attbuf, C, nh, H, W, d);
  else          k_na2d<64><<<tb, 256, 0, s>>>(regA, t.rpb, attbuf, C, nh, H, W, d);
  launch_gemm(s, attbuf, t.proj_w, t.proj_b, ybuf, C, C, HW);
  k_add<<<nb, 256, 0, s>>>(x, ybuf, C * HW);

  // --- FFN branch ---
  k_layernorm<<<pb, 256, 0, s>>>(x, t.n2w, t.n2b, ybuf, C, HW);
  launch_gemm(s, ybuf, t.pin, nullptr, regA, 2 * hf, C, HW);        // u in regA
  int db = (2 * hf * HW + 255) / 256;
  k_dwconv3x3<<<db, 256, 0, s>>>(regA, t.dw, regB, 2 * hf, H, W);   // dw -> regB
  int gb = (hf * HW + 255) / 256;
  k_gate<<<gb, 256, 0, s>>>(regB, regA, hf, HW);                    // g overlays regA
  launch_gemm(s, regA, t.pout, nullptr, ybuf, C, hf, HW);
  k_add<<<nb, 256, 0, s>>>(x, ybuf, C * HW);
}

static void run_stage(hipStream_t s, float* x, const TransP* t2,
                      int C, int nh, int dil, int H, int W, int hf,
                      float* ybuf, float* attbuf, float* regA, float* regB)
{
  trans_block(s, x, t2[0], C, nh, 1,   H, W, hf, ybuf, attbuf, regA, regB);
  trans_block(s, x, t2[1], C, nh, dil, H, W, hf, ybuf, attbuf, regA, regB);
}

extern "C" void kernel_launch(void* const* d_in, const int* in_sizes, int n_in,
                              void* d_out, int out_size, void* d_ws, size_t ws_size,
                              hipStream_t stream)
{
  (void)in_sizes; (void)n_in; (void)out_size; (void)ws_size;
  // ---- parameter walk (setup_inputs dict order) ----
  int ci = 0;
  auto nxt = [&]() { return (const float*)d_in[ci++]; };
  auto readT = [&](TransP& t) {
    t.n1w = nxt(); t.n1b = nxt(); t.qkv_w = nxt(); t.qkv_b = nxt(); t.rpb = nxt();
    t.proj_w = nxt(); t.proj_b = nxt(); t.n2w = nxt(); t.n2b = nxt();
    t.pin = nxt(); t.dw = nxt(); t.pout = nxt();
  };
  const float* inp   = nxt();
  const float* patch = nxt();
  TransP enc1[2]; readT(enc1[0]); readT(enc1[1]);
  const float* down12 = nxt();
  TransP enc2[2]; readT(enc2[0]); readT(enc2[1]);
  const float* down23 = nxt();
  TransP enc3[2]; readT(enc3[0]); readT(enc3[1]);
  const float* down34 = nxt();
  TransP lat[2];  readT(lat[0]);  readT(lat[1]);
  const float* up43 = nxt();
  const float* red3 = nxt();
  TransP dec3[2]; readT(dec3[0]); readT(dec3[1]);
  const float* up32 = nxt();
  const float* red2 = nxt();
  TransP dec2[2]; readT(dec2[0]); readT(dec2[1]);
  const float* up21 = nxt();
  TransP dec1[2]; readT(dec1[0]); readT(dec1[1]);
  TransP refi[2]; readT(refi[0]); readT(refi[1]);
  const float* outw = nxt();

  // ---- workspace arena ----
  char* ws = (char*)d_ws;
  size_t off = 0;
  auto alloc = [&](size_t nf) {
    float* p = (float*)(ws + off);
    off += nf * sizeof(float);
    off = (off + 255) & ~(size_t)255;
    return p;
  };
  const size_t MAXCH = 1048576;   // 64 * 128 * 128 (max C*HW)
  const size_t MAXBG = 5570560;   // 2*170 * 128*128 (max FFN expand)
  float* X   = alloc(MAXCH);
  float* Yb  = alloc(MAXCH);
  float* Att = alloc(MAXCH);
  float* RA  = alloc(MAXBG);
  float* RB  = alloc(MAXBG);
  float* T1  = alloc(MAXCH);
  float* E1  = alloc(524288);     // [32,128,128]
  float* E2  = alloc(262144);     // [64,64,64]
  float* E3  = alloc(131072);     // [128,32,32]

  auto conv = [&](const float* in, const float* w, float* out,
                  int Ci, int Co, int H, int W, const float* add) {
    int n = Co * H * W;
    k_conv3x3<<<(n + 255) / 256, 256, 0, stream>>>(in, w, out, Ci, Co, H, W, add);
  };
  auto d2d = [&](float* dst, const float* src, size_t nf) {
    hipMemcpyAsync(dst, src, nf * sizeof(float), hipMemcpyDeviceToDevice, stream);
  };

  // ---- encoder ----
  conv(inp, patch, X, 3, 32, 128, 128, nullptr);                       // patch embed
  run_stage(stream, X, enc1, 32, 1, 18, 128, 128, 85, Yb, Att, RA, RB);
  d2d(E1, X, 524288);

  conv(X, down12, T1, 32, 16, 128, 128, nullptr);
  k_unshuffle2<<<(262144 + 255) / 256, 256, 0, stream>>>(T1, X, 16, 128, 128);
  run_stage(stream, X, enc2, 64, 2, 9, 64, 64, 170, Yb, Att, RA, RB);
  d2d(E2, X, 262144);

  conv(X, down23, T1, 64, 32, 64, 64, nullptr);
  k_unshuffle2<<<(131072 + 255) / 256, 256, 0, stream>>>(T1, X, 32, 64, 64);
  run_stage(stream, X, enc3, 128, 4, 4, 32, 32, 340, Yb, Att, RA, RB);
  d2d(E3, X, 131072);

  conv(X, down34, T1, 128, 64, 32, 32, nullptr);
  k_unshuffle2<<<(65536 + 255) / 256, 256, 0, stream>>>(T1, X, 64, 32, 32);
  run_stage(stream, X, lat, 256, 8, 2, 16, 16, 680, Yb, Att, RA, RB);

  // ---- decoder ----
  conv(X, up43, RA, 256, 512, 16, 16, nullptr);                        // [512,16,16]
  k_shuffle2<<<(131072 + 255) / 256, 256, 0, stream>>>(RA, T1, 512, 16, 16); // [128,32,32]
  d2d(RA, T1, 131072);
  d2d(RA + 131072, E3, 131072);                                        // concat -> [256,32,32]
  launch_gemm(stream, RA, red3, nullptr, X, 128, 256, 1024);           // 1x1 reduce
  run_stage(stream, X, dec3, 128, 4, 4, 32, 32, 340, Yb, Att, RA, RB);

  conv(X, up32, RA, 128, 256, 32, 32, nullptr);                        // [256,32,32]
  k_shuffle2<<<(262144 + 255) / 256, 256, 0, stream>>>(RA, T1, 256, 32, 32); // [64,64,64]
  d2d(RA, T1, 262144);
  d2d(RA + 262144, E2, 262144);                                        // concat -> [128,64,64]
  launch_gemm(stream, RA, red2, nullptr, X, 64, 128, 4096);
  run_stage(stream, X, dec2, 64, 2, 9, 64, 64, 170, Yb, Att, RA, RB);

  conv(X, up21, RA, 64, 128, 64, 64, nullptr);                         // [128,64,64]
  k_shuffle2<<<(524288 + 255) / 256, 256, 0, stream>>>(RA, T1, 128, 64, 64); // [32,128,128]
  d2d(X, T1, 524288);
  d2d(X + 524288, E1, 524288);                                         // concat -> [64,128,128]
  run_stage(stream, X, dec1, 64, 1, 18, 128, 128, 170, Yb, Att, RA, RB);
  run_stage(stream, X, refi, 64, 1, 18, 128, 128, 170, Yb, Att, RA, RB);

  conv(X, outw, (float*)d_out, 64, 3, 128, 128, inp);                  // output + residual
}
